// WaveNetModel_16449724745415
// MI455X (gfx1250) — matrix operand order repaired and checked by reading, not compiled, not run on hardware
//
#include <hip/hip_runtime.h>
#include <hip/hip_bf16.h>

typedef __attribute__((ext_vector_type(16))) _Float16 v16h;
typedef __attribute__((ext_vector_type(8)))  float    v8f;

#define XSTRIDE 65536
#define NLAYERS 40
#define LDK  136   // layer-kernel LDS row stride (f16 units): 272 B, 16-B aligned
#define LDKE 264   // end-kernel LDS row stride (f16 units): 528 B, 16-B aligned

union H8 {
  uint4 u;
  _Float16 h[8];
};

// ---------------------------------------------------------------------------
// WMMA: D = A(16x32 f16) * B(32x16 f16) + C(16x16 f32), wave32
// ---------------------------------------------------------------------------
__device__ __forceinline__ v8f wmma_f32_f16(v16h a, v16h b, v8f c) {
  return __builtin_amdgcn_wmma_f32_16x16x32_f16(
      false, a, false, b, (short)0, c, false, false);
}

// A fragment from pre-swizzled weights: tile stored [lane][16] contiguous.
__device__ __forceinline__ v16h load_a_sw(const _Float16* __restrict__ Wsw,
                                          int tile, int lane) {
  return *(const v16h*)(Wsw + (((size_t)tile << 5) + lane) * 16);
}

// B fragment from LDS [n][k] tile: 16 contiguous f16 per lane.
// lanes 0-15: N=n0+lane, K=k0..k0+15 ; lanes 16-31: N=n0+lane-16, K=k0+16..k0+31
__device__ __forceinline__ v16h load_b_frag(const _Float16* X, int ldk,
                                            int n0, int k0, int lane) {
  const int n = n0 + (lane & 15);
  const int koff = k0 + ((lane >> 4) << 4);
  return *(const v16h*)(X + n * ldk + koff);
}

// ---------------------------------------------------------------------------
// Weight prep: pack fp32 weights into f16 *fragment-swizzled* layouts.
// Fragment slot s (0..15): K = k0 + 16*(lane>>4) + s
// ---------------------------------------------------------------------------
__device__ __forceinline__ int swz_k(int ks, int lane, int s) {
  return (ks << 5) + ((lane >> 4) << 4) + s;
}

__global__ void prep_fg_kernel(const float* __restrict__ fw,
                               const float* __restrict__ gw,
                               _Float16* __restrict__ dst) {
  const long idx = (long)blockIdx.x * 256 + threadIdx.x;
  if (idx >= (long)NLAYERS * 32 * 512) return;
  const int l = (int)(idx >> 14);
  const int r = (int)(idx & 16383);
  const int tile = r >> 9, lane = (r >> 4) & 31, s = r & 15;
  const int mb = tile >> 2, ks = tile & 3;
  const int m = (mb << 4) + (lane & 15);
  const int kk = swz_k(ks, lane, s);
  const int o = m & 63, tap = kk >> 6, ic = kk & 63;
  const float* src = (m < 64) ? fw : gw;  // [40][64][64][2]
  dst[idx] = (_Float16)src[(((size_t)l * 64 + o) * 64 + ic) * 2 + tap];
}

__global__ void prep_sr_kernel(const float* __restrict__ sw,
                               const float* __restrict__ rw,
                               _Float16* __restrict__ dst) {
  const long idx = (long)blockIdx.x * 256 + threadIdx.x;
  if (idx >= (long)NLAYERS * 40 * 512) return;
  const int l = (int)(idx / 20480);
  const int r = (int)(idx % 20480);
  const int tile = r >> 9, lane = (r >> 4) & 31, s = r & 15;
  const int mb = tile >> 1, ks = tile & 1;
  const int m = (mb << 4) + (lane & 15);
  const int kk = swz_k(ks, lane, s);  // < 64
  const float v = (m < 256) ? sw[((size_t)l * 256 + m) * 64 + kk]
                            : rw[((size_t)l * 64 + (m - 256)) * 64 + kk];
  dst[idx] = (_Float16)v;
}

// generic [M][256] fp32 -> swizzled tiles (mb, ks 0..7)
__global__ void prep_end_kernel(const float* __restrict__ src,
                                _Float16* __restrict__ dst, int ntiles) {
  const int idx = blockIdx.x * 256 + threadIdx.x;
  if (idx >= ntiles * 512) return;
  const int tile = idx >> 9, lane = (idx >> 4) & 31, s = idx & 15;
  const int mb = tile >> 3, ks = tile & 7;
  const int m = (mb << 4) + (lane & 15);
  const int kk = swz_k(ks, lane, s);
  dst[idx] = (_Float16)src[(size_t)m * 256 + kk];
}

// start conv (IN_C==1): xT[t][c] = start_w[c] * input[batch][time], t=time*4+batch
__global__ void start_kernel(const float* __restrict__ in,
                             const float* __restrict__ sw,
                             _Float16* __restrict__ x) {
  const long idx = (long)blockIdx.x * 256 + threadIdx.x;
  if (idx >= 64L * XSTRIDE) return;
  const long t = idx >> 6;
  const int c = (int)(idx & 63);
  x[idx] = (_Float16)(sw[c] * in[(t & 3) * 16384 + (t >> 2)]);
}

// ---------------------------------------------------------------------------
// One WaveNet layer, fused per 128-column tile.
//   x layout: f16 [t][64] ; skip: fp32 [t][256]
// ---------------------------------------------------------------------------
__global__ __launch_bounds__(256) void wavenet_layer_kernel(
    const _Float16* __restrict__ xin, _Float16* __restrict__ xout,
    float* __restrict__ skipT,
    const _Float16* __restrict__ WfgSw,  // 32 tiles (mb0..7 x ks0..3)
    const _Float16* __restrict__ WsrSw,  // 40 tiles (mb0..19 x ks0..1)
    int Tin, int P, int S, int Tout, int skipDrop) {
  __shared__ _Float16 smem[128 * LDK];  // [n][k], 34 KB; xcat then act
  const int tid = threadIdx.x, lane = tid & 31, w = tid >> 5;
  const long t0 = (long)blockIdx.x * 128;

  // Stage xcat: column j, k<64 = x[t0+j], k>=64 = x[t0+j+S]; pad P reads as 0.
  {
    const int j = tid >> 1, half = tid & 1;
    const long t = t0 + j + (half ? (long)S : 0L);
    const long tp = t - P;
    uint4* dst = (uint4*)&smem[j * LDK + (half << 6)];
    if (tp >= 0 && tp < Tin) {
      const uint4* src = (const uint4*)(xin + tp * 64);
#pragma unroll
      for (int g = 0; g < 8; ++g) dst[g] = src[g];
    } else {
      const uint4 z = {0u, 0u, 0u, 0u};
#pragma unroll
      for (int g = 0; g < 8; ++g) dst[g] = z;
    }
  }
  // prefetch next tile (gfx1250 global_prefetch_b8)
  {
    const long tn = t0 + 128 + (tid & 127) - P;
    if (tn >= 0 && tn < Tin) __builtin_prefetch(xin + tn * 64, 0, 1);
  }
  __syncthreads();

  // GEMM1: wave w owns M-block w (fg rows 16w..16w+15), 8 N-blocks, K=128.
  v16h afrag[4];
#pragma unroll
  for (int ks = 0; ks < 4; ++ks) afrag[ks] = load_a_sw(WfgSw, (w << 2) + ks, lane);

  v8f acc[8];
#pragma unroll
  for (int nb = 0; nb < 8; ++nb) {
    v8f c = {};
#pragma unroll
    for (int ks = 0; ks < 4; ++ks)
      c = wmma_f32_f16(afrag[ks], load_b_frag(smem, LDK, nb << 4, ks << 5, lane), c);
    acc[nb] = c;
  }
  __syncthreads();

  // Activation epilogue -> act[n][m] in same LDS (rows 0-63 tanh, 64-127 sigm)
  {
    const int nl = lane & 15;
    const int m0 = (w << 4) + ((lane >> 4) << 3);
#pragma unroll
    for (int nb = 0; nb < 8; ++nb) {
      H8 o;
#pragma unroll
      for (int v = 0; v < 8; ++v) {
        const int m = m0 + v;
        float val = acc[nb][v];
        val = (m < 64) ? tanhf(val) : (1.f / (1.f + __expf(-val)));
        o.h[v] = (_Float16)val;
      }
      *(uint4*)&smem[((nb << 4) + nl) * LDK + m0] = o.u;
    }
  }
  __syncthreads();

  // GEMM2: wave w owns N-block w; B = gated product f*g (v_pk_mul_f16).
  v16h bfrag[2];
  {
    const _Float16* rowp = &smem[((w << 4) + (lane & 15)) * LDK];
    const int hi = (lane >> 4) << 4;
#pragma unroll
    for (int ks = 0; ks < 2; ++ks) {
      const int koff = (ks << 5) + hi;
      v16h f = *(const v16h*)(rowp + koff);
      v16h g = *(const v16h*)(rowp + 64 + koff);
      bfrag[ks] = f * g;
    }
  }

  const int nl = lane & 15;
  const int hi8 = (lane >> 4) << 3;
  const long tcol = t0 + (w << 4) + nl;
  for (int mb = 0; mb < 20; ++mb) {  // 320 rows = 256 skip + 64 res
    v8f c = {};
#pragma unroll
    for (int ks = 0; ks < 2; ++ks)
      c = wmma_f32_f16(load_a_sw(WsrSw, (mb << 1) + ks, lane), bfrag[ks], c);
    if (tcol < Tout) {
      const int m0 = (mb << 4) + hi8;
      if (m0 < 256) {  // skip accumulation, tail-aligned
        const long u = tcol - skipDrop;
        if (u >= 0) {
          float4* sp = (float4*)(skipT + u * 256 + m0);
          float4 s0 = sp[0], s1 = sp[1];
          s0.x += c[0]; s0.y += c[1]; s0.z += c[2]; s0.w += c[3];
          s1.x += c[4]; s1.y += c[5]; s1.z += c[6]; s1.w += c[7];
          sp[0] = s0; sp[1] = s1;
        }
      } else {  // residual: res(h) + x_old[t+S]
        const int cc0 = m0 - 256;
        H8 old, res;
        old.u = *(const uint4*)(xin + (tcol + S - P) * 64 + cc0);
#pragma unroll
        for (int v = 0; v < 8; ++v)
          res.h[v] = (_Float16)(c[v] + (float)old.h[v]);
        *(uint4*)(xout + tcol * 64 + cc0) = res.u;
      }
    }
  }
}

// ---------------------------------------------------------------------------
// End head: relu(skip) -> end1(256x256)+b relu -> end2(128x256)+b.
// skip: fp32 [t][256] ; y: fp32 [t][128]
// ---------------------------------------------------------------------------
__global__ __launch_bounds__(256) void wavenet_end_kernel(
    const float* __restrict__ skipT,
    const _Float16* __restrict__ W1Sw, const float* __restrict__ b1,
    const _Float16* __restrict__ W2Sw, const float* __restrict__ b2,
    float* __restrict__ y, int Tfinal) {
  __shared__ _Float16 smem[64 * LDKE];  // [n][k], 33 KB; relu(skip) then e1
  const int tid = threadIdx.x, lane = tid & 31, w = tid >> 5;
  const long t0 = (long)blockIdx.x * 64;

  // stage relu(skip) as f16 [n][256]
  {
    const int j = tid >> 2, q = tid & 3;
    const long t = t0 + j;
    _Float16* dst = &smem[j * LDKE + (q << 6)];
    if (t < Tfinal) {
      const float4* src = (const float4*)(skipT + t * 256 + (q << 6));
#pragma unroll
      for (int g = 0; g < 16; ++g) {
        const float4 f = src[g];
        H8 o;  // pack one float4 into a 4xf16 (8-B) store
        o.h[0] = (_Float16)fmaxf(f.x, 0.f);
        o.h[1] = (_Float16)fmaxf(f.y, 0.f);
        o.h[2] = (_Float16)fmaxf(f.z, 0.f);
        o.h[3] = (_Float16)fmaxf(f.w, 0.f);
        *(uint2*)(dst + (g << 2)) = make_uint2(o.u.x, o.u.y);
      }
    } else {
      const uint2 z = make_uint2(0u, 0u);
#pragma unroll
      for (int g = 0; g < 16; ++g) *(uint2*)(dst + (g << 2)) = z;
    }
  }
  __syncthreads();

  // end1: M=256 (wave w -> mb 2w,2w+1), N=64 (4 nb), K=256 (8 ksteps)
  v8f acc[8];
#pragma unroll
  for (int i = 0; i < 8; ++i) {
    const int mb = (w << 1) + (i >> 2);
    const int nb = i & 3;
    v8f c = {};
#pragma unroll
    for (int ks = 0; ks < 8; ++ks)
      c = wmma_f32_f16(load_a_sw(W1Sw, (mb << 3) + ks, lane),
                       load_b_frag(smem, LDKE, nb << 4, ks << 5, lane), c);
    acc[i] = c;
  }
  __syncthreads();

  {  // bias + relu -> restage e1 into same LDS [n][m]
    const int nl = lane & 15, hi8 = (lane >> 4) << 3;
#pragma unroll
    for (int i = 0; i < 8; ++i) {
      const int mb = (w << 1) + (i >> 2);
      const int nb = i & 3;
      const int m0 = (mb << 4) + hi8;
      H8 o;
#pragma unroll
      for (int v = 0; v < 8; ++v)
        o.h[v] = (_Float16)fmaxf(acc[i][v] + b1[m0 + v], 0.f);
      *(uint4*)&smem[((nb << 4) + nl) * LDKE + m0] = o.u;
    }
  }
  __syncthreads();

  {  // end2: M=128 (wave w -> mb w), K=256
    const int nl = lane & 15, hi8 = (lane >> 4) << 3;
#pragma unroll
    for (int nb = 0; nb < 4; ++nb) {
      v8f c = {};
#pragma unroll
      for (int ks = 0; ks < 8; ++ks)
        c = wmma_f32_f16(load_a_sw(W2Sw, (w << 3) + ks, lane),
                         load_b_frag(smem, LDKE, nb << 4, ks << 5, lane), c);
      const long t = t0 + (nb << 4) + nl;
      if (t < Tfinal) {
        const int m0 = (w << 4) + hi8;
        float4* yp = (float4*)(y + t * 128 + m0);
        yp[0] = make_float4(c[0] + b2[m0 + 0], c[1] + b2[m0 + 1],
                            c[2] + b2[m0 + 2], c[3] + b2[m0 + 3]);
        yp[1] = make_float4(c[4] + b2[m0 + 4], c[5] + b2[m0 + 5],
                            c[6] + b2[m0 + 6], c[7] + b2[m0 + 7]);
      }
    }
  }
}

// Adaptive avg-pool along time (channel pool 128->128 identity).
__global__ void wavenet_pool_kernel(const float* __restrict__ y,
                                    float* __restrict__ out, int Tfinal) {
  const int bc = blockIdx.x;
  const int i = bc >> 7, c = bc & 127;
  const int p = threadIdx.x;
  const int L = Tfinal >> 2;  // 14336
  const int s = (p * L) >> 7;
  const int e = ((p + 1) * L + 127) >> 7;
  float acc = 0.f;
  for (int j = s; j < e; ++j)
    acc += y[(size_t)((j << 2) + i) * 128 + c];
  out[(((size_t)i << 7) + c) * 128 + p] = acc / (float)(e - s);
}

// ---------------------------------------------------------------------------
extern "C" void kernel_launch(void* const* d_in, const int* in_sizes, int n_in,
                              void* d_out, int out_size, void* d_ws,
                              size_t ws_size, hipStream_t stream) {
  (void)in_sizes; (void)n_in; (void)out_size; (void)ws_size;
  const float* input    = (const float*)d_in[0];
  const float* start_w  = (const float*)d_in[1];
  const float* filter_w = (const float*)d_in[2];
  const float* gate_w   = (const float*)d_in[3];
  const float* res_w    = (const float*)d_in[4];
  const float* skip_w   = (const float*)d_in[5];
  const float* end1_w   = (const float*)d_in[6];
  const float* end1_b   = (const float*)d_in[7];
  const float* end2_w   = (const float*)d_in[8];
  const float* end2_b   = (const float*)d_in[9];
  float* out = (float*)d_out;

  // Per-layer flat geometry, replicating dilate()'s left-padding exactly.
  int T = 65536;
  int S[NLAYERS], P[NLAYERS], TIN[NLAYERS], TOUT[NLAYERS];
  for (int i = 0; i < NLAYERS; ++i) {
    const int d = 1 << (i % 10);
    const int init_d = (i % 10 == 0) ? ((i == 0) ? 1 : 512) : (d >> 1);
    S[i] = 4 * d;
    int pad = 0;
    if (d > init_d) {
      const int n_old = 4 * init_d;
      const int l_old = T / n_old;
      const int f = d / init_d;
      const int new_l = ((l_old + f - 1) / f) * f;
      pad = (new_l - l_old) * n_old;
    }
    P[i] = pad;
    TIN[i] = T;
    TOUT[i] = T + pad - S[i];
    T = TOUT[i];
  }
  const int Tfinal = T;  // 57344 -> pooled length 14336 = 112*128

  // Workspace layout
  char* ws = (char*)d_ws;
  size_t off = 0;
  auto wsalloc = [&](size_t bytes) -> void* {
    void* p = ws + off;
    off += (bytes + 255) & ~(size_t)255;
    return p;
  };
  _Float16* xA = (_Float16*)wsalloc(sizeof(_Float16) * 64 * (size_t)XSTRIDE);
  _Float16* xB = (_Float16*)wsalloc(sizeof(_Float16) * 64 * (size_t)XSTRIDE);
  float* skip = (float*)wsalloc(sizeof(float) * 256 * (size_t)Tfinal);
  float* yb = (float*)wsalloc(sizeof(float) * 128 * (size_t)Tfinal);
  _Float16* WfgSw = (_Float16*)wsalloc(sizeof(_Float16) * NLAYERS * 32 * 512);
  _Float16* WsrSw = (_Float16*)wsalloc(sizeof(_Float16) * NLAYERS * 40 * 512);
  _Float16* W1Sw  = (_Float16*)wsalloc(sizeof(_Float16) * 128 * 512);
  _Float16* W2Sw  = (_Float16*)wsalloc(sizeof(_Float16) * 64 * 512);

  prep_fg_kernel<<<(NLAYERS * 32 * 512 + 255) / 256, 256, 0, stream>>>(
      filter_w, gate_w, WfgSw);
  prep_sr_kernel<<<(NLAYERS * 40 * 512 + 255) / 256, 256, 0, stream>>>(
      skip_w, res_w, WsrSw);
  prep_end_kernel<<<(128 * 512 + 255) / 256, 256, 0, stream>>>(end1_w, W1Sw, 128);
  prep_end_kernel<<<(64 * 512 + 255) / 256, 256, 0, stream>>>(end2_w, W2Sw, 64);
  start_kernel<<<(64 * XSTRIDE) / 256, 256, 0, stream>>>(input, start_w, xA);
  hipMemsetAsync(skip, 0, sizeof(float) * 256 * (size_t)Tfinal, stream);

  _Float16* xin = xA;
  _Float16* xout = xB;
  for (int i = 0; i < NLAYERS; ++i) {
    const int blocks = (TOUT[i] + 127) / 128;
    wavenet_layer_kernel<<<blocks, 256, 0, stream>>>(
        xin, xout, skip, WfgSw + (size_t)i * 32 * 512,
        WsrSw + (size_t)i * 40 * 512, TIN[i], P[i], S[i], TOUT[i],
        TOUT[i] - Tfinal);
    _Float16* tmp = xin; xin = xout; xout = tmp;
  }

  wavenet_end_kernel<<<(Tfinal + 63) / 64, 256, 0, stream>>>(
      skip, W1Sw, end1_b, W2Sw, end2_b, yb, Tfinal);
  wavenet_pool_kernel<<<4 * 128, 128, 0, stream>>>(yb, out, Tfinal);
}
